// GraphNeuralODEFuncHierarchical_58884001628479
// MI455X (gfx1250) — matrix operand order repaired and checked by reading, not compiled, not run on hardware
//
#include <hip/hip_runtime.h>
#include <hip/hip_bf16.h>
#include <stdint.h>

// ---------------------------------------------------------------------------
// Shapes: B=4, N_p=32768, N_o=M=64, D=13, H=256. Edge rows R = B*64*64 = 16384.
// dx_p = d_out[0 .. 4*32768*13), dx_o = d_out[4*32768*13 .. +4*64*13)
// ---------------------------------------------------------------------------

typedef __attribute__((ext_vector_type(16))) _Float16 v16h;
typedef __attribute__((ext_vector_type(8)))  float    v8f;

union U16h { uint32_t u[8]; v16h v; };

// ---------------------------------------------------------------------------
// Generic WMMA GEMM:  C[M x ldc] = act( A[M x Kpad] * Bt[N x Kpad]^T + bias )
// A, Bt are f16 row-major; Kpad % 32 == 0; M % 16 == 0.
// One wave (32 threads) computes one 16x16 output tile via v_wmma_f32_16x16x32_f16.
// VGPR layouts used here (A/B in contiguous k order, same for both; not the ISA table order):
//   A (16x32 f16): lanes 0-15 -> M=lane, V0..7 hold K pairs 0..15
//                  lanes16-31 -> M=lane-16, V0..7 hold 16..31
//   B (32x16 f16): V0..7 : lanes 0-15 N=lane  K pairs 0..15 ; lanes16-31 K 16..31
//   C/D f32: reg r -> row = r + 8*(lane>=16), col = lane&15
// ---------------------------------------------------------------------------
template<bool GELU_, bool OUTF16>
__global__ __launch_bounds__(32)
void gemm_wmma(const _Float16* __restrict__ A, const _Float16* __restrict__ Bt,
               const float* __restrict__ bias, void* __restrict__ C,
               int Kpad, int Nact, int ntn, int ldc)
{
    int mt   = blockIdx.x / ntn;
    int nt   = blockIdx.x % ntn;
    int lane = threadIdx.x;
    int l15  = lane & 15;
    int hi   = lane >> 4;                 // 0 or 1 (lane half)
    int kb   = hi * 8;

    const _Float16* ap = A  + (size_t)(mt * 16 + l15) * Kpad;
    const _Float16* bp = Bt + (size_t)(nt * 16 + l15) * Kpad;

    v8f acc = {};
    for (int k0 = 0; k0 < Kpad; k0 += 32) {
        U16h a, b;
#pragma unroll
        for (int v = 0; v < 4; ++v) {
            a.u[v]     = *(const uint32_t*)(ap + k0 + hi * 16 + 2 * v);
            a.u[v + 4] = *(const uint32_t*)(ap + k0 + hi * 16 + 8 + 2 * v);
        }
#pragma unroll
        for (int v = 0; v < 8; ++v)
            b.u[v] = *(const uint32_t*)(bp + k0 + hi * 16 + 2 * v);
        acc = __builtin_amdgcn_wmma_f32_16x16x32_f16(false, a.v, false, b.v,
                                                     (short)0, acc, false, false);
    }

    int col = nt * 16 + l15;
    if (col < Nact) {
        float bv = bias ? bias[col] : 0.0f;
#pragma unroll
        for (int r = 0; r < 8; ++r) {
            int   row = mt * 16 + r + hi * 8;
            float x   = acc[r] + bv;
            if (GELU_) x = 0.5f * x * (1.0f + erff(x * 0.70710678118654752f));
            if (OUTF16) ((_Float16*)C)[(size_t)row * ldc + col] = (_Float16)x;
            else        ((float*)   C)[(size_t)row * ldc + col] = x;
        }
    }
}

// W [din x dout] f32 -> Wt [Nrows x Kpad] f16 (transposed, zero padded)
__global__ void prep_wt(const float* __restrict__ W, _Float16* __restrict__ Wt,
                        int din, int dout, int Kpad, int Nrows)
{
    int idx = blockIdx.x * 256 + threadIdx.x;
    if (idx >= Nrows * Kpad) return;
    int n = idx / Kpad, k = idx % Kpad;
    float v = (n < dout && k < din) ? W[k * dout + n] : 0.0f;
    Wt[idx] = (_Float16)v;
}

// Edge features [h_i(13), h_j(13), diff(3), dist(1), t(1), 0] -> f16 [16384 x 32]
// plus un-normalized adjacency weight A[r] = exp(-d2/sigma^2) (diag = 1 naturally)
__global__ void edge_input_kernel(const float* __restrict__ x_o,
                                  const float* __restrict__ tptr,
                                  _Float16* __restrict__ X1, float* __restrict__ Aw)
{
    int r = blockIdx.x * 256 + threadIdx.x;
    if (r >= 16384) return;
    int b = r >> 12, rem = r & 4095, i = rem >> 6, j = rem & 63;
    const float* xi = x_o + ((size_t)b * 64 + i) * 13;
    const float* xj = x_o + ((size_t)b * 64 + j) * 13;
    float dx = xj[0] - xi[0], dy = xj[1] - xi[1], dz = xj[2] - xi[2];
    float d2 = dx * dx + dy * dy + dz * dz;
    Aw[r] = expf(-d2 / 0.09f);                // sigma = 0.3
    _Float16* o = X1 + (size_t)r * 32;
#pragma unroll
    for (int k = 0; k < 13; ++k) o[k]      = (_Float16)xi[k];
#pragma unroll
    for (int k = 0; k < 13; ++k) o[13 + k] = (_Float16)xj[k];
    o[26] = (_Float16)dx; o[27] = (_Float16)dy; o[28] = (_Float16)dz;
    o[29] = (_Float16)sqrtf(d2);
    o[30] = (_Float16)(*tptr);
    o[31] = (_Float16)0.0f;
}

// m_obj[n,d] = sum_j A[n,j]*msg[n*64+j,d] / sum_j A[n,j]   (n = b*64+i, 256 nodes)
__global__ void aggregate_kernel(const float* __restrict__ MSG,
                                 const float* __restrict__ Aw,
                                 float* __restrict__ Mobj)
{
    int idx = blockIdx.x * 256 + threadIdx.x;
    if (idx >= 256 * 13) return;
    int n = idx / 13, d = idx % 13;
    const float* arow = Aw + n * 64;
    const float* mrow = MSG + (size_t)n * 64 * 16 + d;
    float s = 0.0f, acc = 0.0f;
    for (int j = 0; j < 64; ++j) { float a = arow[j]; s += a; acc += a * mrow[(size_t)j * 16]; }
    Mobj[idx] = acc / s;
}

// node_in = [x_o(13), m_obj(13), t(1), pad] -> f16 [256 x 32]
__global__ void node_input_kernel(const float* __restrict__ x_o,
                                  const float* __restrict__ Mobj,
                                  const float* __restrict__ tptr,
                                  _Float16* __restrict__ NIN)
{
    int n = blockIdx.x * 64 + threadIdx.x;
    if (n >= 256) return;
    _Float16* o = NIN + (size_t)n * 32;
    const float* x = x_o + (size_t)n * 13;
    const float* m = Mobj + (size_t)n * 13;
#pragma unroll
    for (int k = 0; k < 13; ++k) o[k]      = (_Float16)x[k];
#pragma unroll
    for (int k = 0; k < 13; ++k) o[13 + k] = (_Float16)m[k];
    o[26] = (_Float16)(*tptr);
#pragma unroll
    for (int k = 27; k < 32; ++k) o[k] = (_Float16)0.0f;
}

// dx_o = [vel(3), dv(3), 0.5*quat_mul((0,omega), q)(4), dw(3)]
__global__ void dxo_kernel(const float* __restrict__ x_o,
                           const float* __restrict__ DV,
                           const float* __restrict__ DW,
                           float* __restrict__ out)
{
    int n = blockIdx.x * 64 + threadIdx.x;
    if (n >= 256) return;
    const float* x = x_o + (size_t)n * 13;
    float qw = x[6], qx = x[7], qy = x[8], qz = x[9];
    float wx = x[10], wy = x[11], wz = x[12];
    float dqw = 0.5f * (-(wx * qx + wy * qy + wz * qz));
    float dqx = 0.5f * ( wx * qw + wy * qz - wz * qy);
    float dqy = 0.5f * ( wy * qw - wx * qz + wz * qx);
    float dqz = 0.5f * ( wz * qw + wx * qy - wy * qx);
    float* o = out + (size_t)n * 13;
    o[0] = x[3]; o[1] = x[4]; o[2] = x[5];
    o[3] = DV[n * 16 + 0]; o[4] = DV[n * 16 + 1]; o[5] = DV[n * 16 + 2];
    o[6] = dqw; o[7] = dqx; o[8] = dqy; o[9] = dqz;
    o[10] = DW[n * 16 + 0]; o[11] = DW[n * 16 + 1]; o[12] = DW[n * 16 + 2];
}

// Particle kernel: final_vel = (sum_o S*M_o)·pos + sum_o S*c_o
// M_o = [w]x R_o (unnormalized quat sandwich), c_o = vel_o - M_o·COM_o
__global__ __launch_bounds__(256)
void particle_kernel(const float* __restrict__ x_p, const float* __restrict__ x_o,
                     const float* __restrict__ S, float* __restrict__ out)
{
    __shared__ float sObj[64 * 12];
    __shared__ float sS[256 * 65];            // padded rows: bank-conflict free
    int b  = blockIdx.x >> 7;                 // 128 blocks per batch
    int p0 = (blockIdx.x & 127) << 8;
    int t  = threadIdx.x;

    if (t < 64) {
        const float* x = x_o + ((size_t)b * 64 + t) * 13;
        float cx = x[0], cy = x[1], cz = x[2];
        float vx = x[3], vy = x[4], vz = x[5];
        float qw = x[6], qx = x[7], qy = x[8], qz = x[9];
        float wx = x[10], wy = x[11], wz = x[12];
        float s2 = qw * qw - (qx * qx + qy * qy + qz * qz);
        float R00 = s2 + 2.f*qx*qx, R01 = 2.f*qx*qy - 2.f*qw*qz, R02 = 2.f*qx*qz + 2.f*qw*qy;
        float R10 = 2.f*qx*qy + 2.f*qw*qz, R11 = s2 + 2.f*qy*qy, R12 = 2.f*qy*qz - 2.f*qw*qx;
        float R20 = 2.f*qx*qz - 2.f*qw*qy, R21 = 2.f*qy*qz + 2.f*qw*qx, R22 = s2 + 2.f*qz*qz;
        float M00 = -wz*R10 + wy*R20, M01 = -wz*R11 + wy*R21, M02 = -wz*R12 + wy*R22;
        float M10 =  wz*R00 - wx*R20, M11 =  wz*R01 - wx*R21, M12 =  wz*R02 - wx*R22;
        float M20 = -wy*R00 + wx*R10, M21 = -wy*R01 + wx*R11, M22 = -wy*R02 + wx*R12;
        float* d = sObj + t * 12;
        d[0] = M00; d[1] = M01; d[2] = M02;
        d[3] = M10; d[4] = M11; d[5] = M12;
        d[6] = M20; d[7] = M21; d[8] = M22;
        d[9]  = vx - (M00 * cx + M01 * cy + M02 * cz);
        d[10] = vy - (M10 * cx + M11 * cy + M12 * cz);
        d[11] = vz - (M20 * cx + M21 * cy + M22 * cz);
    }

    const float4* S4 = (const float4*)(S + ((size_t)b * 32768 + p0) * 64);
#pragma unroll
    for (int it = 0; it < 16; ++it) {
        int idx = it * 256 + t;
        float4 v = S4[idx];
        int L = idx << 2, pl = L >> 6, oo = L & 63;
        float* d = sS + pl * 65 + oo;
        d[0] = v.x; d[1] = v.y; d[2] = v.z; d[3] = v.w;
    }
    __syncthreads();

    size_t prow = (size_t)b * 32768 + p0 + t;
    const float* xp = x_p + prow * 13;
    float px = xp[10], py = xp[11], pz = xp[12];
    float cvx = xp[3], cvy = xp[4], cvz = xp[5];
    float am0=0,am1=0,am2=0,am3=0,am4=0,am5=0,am6=0,am7=0,am8=0,ac0=0,ac1=0,ac2=0;
    const float* srow = sS + t * 65;
#pragma unroll 4
    for (int o2 = 0; o2 < 64; ++o2) {
        float s = srow[o2];
        const float* d = sObj + o2 * 12;
        am0 += s * d[0]; am1 += s * d[1]; am2 += s * d[2];
        am3 += s * d[3]; am4 += s * d[4]; am5 += s * d[5];
        am6 += s * d[6]; am7 += s * d[7]; am8 += s * d[8];
        ac0 += s * d[9]; ac1 += s * d[10]; ac2 += s * d[11];
    }
    float fx = am0 * px + am1 * py + am2 * pz + ac0;
    float fy = am3 * px + am4 * py + am5 * pz + ac1;
    float fz = am6 * px + am7 * py + am8 * pz + ac2;
    float* o = out + prow * 13;
    o[0] = fx; o[1] = fy; o[2] = fz;
    o[3] = fx - cvx; o[4] = fy - cvy; o[5] = fz - cvz;
    o[6] = 0.f; o[7] = 0.f; o[8] = 0.f; o[9] = 0.f;
    o[10] = 0.f; o[11] = 0.f; o[12] = 0.f;
}

// ---------------------------------------------------------------------------
extern "C" void kernel_launch(void* const* d_in, const int* in_sizes, int n_in,
                              void* d_out, int out_size, void* d_ws, size_t ws_size,
                              hipStream_t stream)
{
    const float* t   = (const float*)d_in[0];
    const float* x_p = (const float*)d_in[1];
    const float* x_o = (const float*)d_in[2];
    const float* S   = (const float*)d_in[3];
    const float* ew0 = (const float*)d_in[4];  const float* eb0 = (const float*)d_in[5];
    const float* ew1 = (const float*)d_in[6];  const float* eb1 = (const float*)d_in[7];
    const float* ew2 = (const float*)d_in[8];  const float* eb2 = (const float*)d_in[9];
    const float* vw0 = (const float*)d_in[10]; const float* vb0 = (const float*)d_in[11];
    const float* vw1 = (const float*)d_in[12]; const float* vb1 = (const float*)d_in[13];
    const float* vw2 = (const float*)d_in[14]; const float* vb2 = (const float*)d_in[15];
    const float* vw3 = (const float*)d_in[16]; const float* vb3 = (const float*)d_in[17];
    const float* ow0 = (const float*)d_in[18]; const float* ob0 = (const float*)d_in[19];
    const float* ow1 = (const float*)d_in[20]; const float* ob1 = (const float*)d_in[21];
    const float* ow2 = (const float*)d_in[22]; const float* ob2 = (const float*)d_in[23];
    const float* ow3 = (const float*)d_in[24]; const float* ob3 = (const float*)d_in[25];

    char* ws = (char*)d_ws;
    size_t off = 0;
    auto alloc = [&](size_t bytes) -> char* {
        char* p = ws + off;
        off = (off + bytes + 255) & ~(size_t)255;
        return p;
    };

    _Float16* ew0t = (_Float16*)alloc(256 * 32 * 2);
    _Float16* ew1t = (_Float16*)alloc(256 * 256 * 2);
    _Float16* ew2t = (_Float16*)alloc(16 * 256 * 2);
    _Float16* vw0t = (_Float16*)alloc(256 * 32 * 2);
    _Float16* vw1t = (_Float16*)alloc(256 * 256 * 2);
    _Float16* vw2t = (_Float16*)alloc(256 * 256 * 2);
    _Float16* vw3t = (_Float16*)alloc(16 * 256 * 2);
    _Float16* ow0t = (_Float16*)alloc(256 * 32 * 2);
    _Float16* ow1t = (_Float16*)alloc(256 * 256 * 2);
    _Float16* ow2t = (_Float16*)alloc(256 * 256 * 2);
    _Float16* ow3t = (_Float16*)alloc(16 * 256 * 2);
    _Float16* X1   = (_Float16*)alloc((size_t)16384 * 32 * 2);
    _Float16* H1   = (_Float16*)alloc((size_t)16384 * 256 * 2);
    _Float16* H2   = (_Float16*)alloc((size_t)16384 * 256 * 2);
    float*    MSG  = (float*)   alloc((size_t)16384 * 16 * 4);
    float*    Aw   = (float*)   alloc((size_t)16384 * 4);
    float*    Mobj = (float*)   alloc((size_t)256 * 13 * 4);
    _Float16* NIN  = (_Float16*)alloc(256 * 32 * 2);
    _Float16* NH1  = (_Float16*)alloc(256 * 256 * 2);
    _Float16* NH2  = (_Float16*)alloc(256 * 256 * 2);
    float*    DV   = (float*)   alloc(256 * 16 * 4);
    float*    DW   = (float*)   alloc(256 * 16 * 4);

    auto prep = [&](const float* W, _Float16* Wt, int din, int dout, int Kpad, int Nr) {
        int tot = Nr * Kpad;
        prep_wt<<<(tot + 255) / 256, 256, 0, stream>>>(W, Wt, din, dout, Kpad, Nr);
    };
    prep(ew0, ew0t, 31, 256, 32, 256);
    prep(ew1, ew1t, 256, 256, 256, 256);
    prep(ew2, ew2t, 256, 13, 256, 16);
    prep(vw0, vw0t, 27, 256, 32, 256);
    prep(vw1, vw1t, 256, 256, 256, 256);
    prep(vw2, vw2t, 256, 256, 256, 256);
    prep(vw3, vw3t, 256, 3, 256, 16);
    prep(ow0, ow0t, 27, 256, 32, 256);
    prep(ow1, ow1t, 256, 256, 256, 256);
    prep(ow2, ow2t, 256, 256, 256, 256);
    prep(ow3, ow3t, 256, 3, 256, 16);

    // ---- edge pipeline (16384 rows) ----
    edge_input_kernel<<<64, 256, 0, stream>>>(x_o, t, X1, Aw);
    gemm_wmma<true,  true ><<<1024 * 16, 32, 0, stream>>>(X1, ew0t, eb0, H1, 32, 256, 16, 256);
    gemm_wmma<true,  true ><<<1024 * 16, 32, 0, stream>>>(H1, ew1t, eb1, H2, 256, 256, 16, 256);
    gemm_wmma<false, false><<<1024 * 1, 32, 0, stream>>>(H2, ew2t, eb2, MSG, 256, 13, 1, 16);
    aggregate_kernel<<<13, 256, 0, stream>>>(MSG, Aw, Mobj);

    // ---- node pipeline (256 rows) ----
    node_input_kernel<<<4, 64, 0, stream>>>(x_o, Mobj, t, NIN);
    gemm_wmma<true,  true ><<<16 * 16, 32, 0, stream>>>(NIN, vw0t, vb0, NH1, 32, 256, 16, 256);
    gemm_wmma<true,  true ><<<16 * 16, 32, 0, stream>>>(NH1, vw1t, vb1, NH2, 256, 256, 16, 256);
    gemm_wmma<true,  true ><<<16 * 16, 32, 0, stream>>>(NH2, vw2t, vb2, NH1, 256, 256, 16, 256);
    gemm_wmma<false, false><<<16 * 1, 32, 0, stream>>>(NH1, vw3t, vb3, DV, 256, 3, 1, 16);
    gemm_wmma<true,  true ><<<16 * 16, 32, 0, stream>>>(NIN, ow0t, ob0, NH1, 32, 256, 16, 256);
    gemm_wmma<true,  true ><<<16 * 16, 32, 0, stream>>>(NH1, ow1t, ob1, NH2, 256, 256, 16, 256);
    gemm_wmma<true,  true ><<<16 * 16, 32, 0, stream>>>(NH2, ow2t, ob2, NH1, 256, 256, 16, 256);
    gemm_wmma<false, false><<<16 * 1, 32, 0, stream>>>(NH1, ow3t, ob3, DW, 256, 3, 1, 16);

    float* out = (float*)d_out;
    dxo_kernel<<<4, 64, 0, stream>>>(x_o, DV, DW, out + (size_t)4 * 32768 * 13);

    // ---- particle pipeline (memory-bound, independent of MLPs) ----
    particle_kernel<<<512, 256, 0, stream>>>(x_p, x_o, S, out);
}
